// OuterProductMean_57363583205596
// MI455X (gfx1250) — matrix operand order repaired and checked by reading, not compiled, not run on hardware
//
#include <hip/hip_runtime.h>

typedef __attribute__((ext_vector_type(16))) __bf16 v16bf;
typedef __attribute__((ext_vector_type(8)))  __bf16 v8bf;
typedef __attribute__((ext_vector_type(8)))  float  v8f;

#define NSEQ 512
#define NRES 256
#define CM   256
#define CO   32
#define CF   128

static __device__ __forceinline__ v16bf pack16(v8bf lo, v8bf hi) {
  v16bf r;
#pragma unroll
  for (int i = 0; i < 8; ++i) { r[i] = lo[i]; r[i + 8] = hi[i]; }
  return r;
}

// ---------------------------------------------------------------------------
// output_w [32,32,128] f32  ->  W2_t [f][ce] bf16 (B-operand layout, N-major)
// ---------------------------------------------------------------------------
__global__ __launch_bounds__(256) void k_w2t(const float* __restrict__ ow,
                                             __bf16* __restrict__ w2t) {
  int idx = blockIdx.x * 256 + threadIdx.x;     // 128*1024 = 131072
  int f  = idx >> 10;
  int ce = idx & 1023;
  w2t[idx] = (__bf16)ow[ce * CF + f];
}

// ---------------------------------------------------------------------------
// norm[b][d] = sum_a mask[a][b] * mask[a][d]
// ---------------------------------------------------------------------------
__global__ __launch_bounds__(256) void k_norm(const float* __restrict__ mask,
                                              float* __restrict__ nrm) {
  int b = blockIdx.x;
  int d = threadIdx.x;
  float s = 0.f;
  for (int a = 0; a < NSEQ; ++a) s += mask[a * NRES + b] * mask[a * NRES + d];
  nrm[b * NRES + d] = s;
}

// ---------------------------------------------------------------------------
// LayerNorm + left/right projections, fused via WMMA.
// Each wave owns 16 flattened rows r = a*NRES + b.
// Writes left_t[b][c][a], right_t[d][e][a] (bf16, K(=a)-contiguous).
// ---------------------------------------------------------------------------
__global__ __launch_bounds__(256) void k_ln_lr(
    const float* __restrict__ act,  const float* __restrict__ mask,
    const float* __restrict__ ln_s, const float* __restrict__ ln_o,
    const float* __restrict__ lw,   const float* __restrict__ lb,
    const float* __restrict__ rw,   const float* __restrict__ rb,
    __bf16* __restrict__ left_t,    __bf16* __restrict__ right_t) {
  __shared__ alignas(16) __bf16 wlds[64 * 272];   // 64 cols (32 left | 32 right), col-major, padded
  const int tid  = threadIdx.x;
  const int w    = tid >> 5;
  const int lane = tid & 31;
  const int hf   = lane >> 4;
  const int l16  = lane & 15;

  // Stage weights to LDS as bf16, column-major (stride 272 to spread banks).
  for (int idx = tid; idx < 64 * 256; idx += 256) {
    int col = idx & 63, k = idx >> 6;
    float v = (col < 32) ? lw[k * CO + col] : rw[k * CO + (col - 32)];
    wlds[col * 272 + k] = (__bf16)v;
  }

  const int r0 = blockIdx.x * 128 + w * 16;

  // Per-row mean / rstd (cross-lane butterfly); each lane keeps row m == l16.
  float my_mu = 0.f, my_rs = 0.f;
  for (int rr = 0; rr < 16; ++rr) {
    const float* rp = act + (size_t)(r0 + rr) * CM;
    float s = 0.f, q = 0.f;
#pragma unroll
    for (int i = 0; i < 8; ++i) { float v = rp[lane + 32 * i]; s += v; q += v * v; }
#pragma unroll
    for (int off = 16; off > 0; off >>= 1) {
      s += __shfl_xor(s, off, 32);
      q += __shfl_xor(q, off, 32);
    }
    float mu  = s * (1.f / CM);
    float var = q * (1.f / CM) - mu * mu;
    float rs  = rsqrtf(var + 1e-5f);
    if (rr == l16) { my_mu = mu; my_rs = rs; }
  }
  __syncthreads();

  // [16 x 256] @ [256 x 64] via 8 k-steps of 32; n-tiles: 0,1=left 2,3=right.
  v8f acc[4] = {};
  const float* xrow = act + (size_t)(r0 + l16) * CM;
#pragma unroll
  for (int kk = 0; kk < 8; ++kk) {
    const int k0 = kk * 32;
    v8f x0 = *(const v8f*)(xrow + k0 + hf * 8);
    v8f x1 = *(const v8f*)(xrow + k0 + 16 + hf * 8);
    v8f s0 = *(const v8f*)(ln_s + k0 + hf * 8);
    v8f s1 = *(const v8f*)(ln_s + k0 + 16 + hf * 8);
    v8f o0 = *(const v8f*)(ln_o + k0 + hf * 8);
    v8f o1 = *(const v8f*)(ln_o + k0 + 16 + hf * 8);
    v16bf af;
#pragma unroll
    for (int i = 0; i < 8; ++i) {
      af[i]     = (__bf16)((x0[i] - my_mu) * my_rs * s0[i] + o0[i]);
      af[i + 8] = (__bf16)((x1[i] - my_mu) * my_rs * s1[i] + o1[i]);
    }
#pragma unroll
    for (int t = 0; t < 4; ++t) {
      const __bf16* bp = wlds + (t * 16 + l16) * 272 + k0 + hf * 8;
      v16bf bfr = pack16(*(const v8bf*)bp, *(const v8bf*)(bp + 16));
      acc[t] = __builtin_amdgcn_wmma_f32_16x16x32_bf16(
          false, af, false, bfr, (short)0, acc[t], false, false);
    }
  }

  // Epilogue: + bias, * mask[row], store transposed bf16.
#pragma unroll
  for (int t = 0; t < 4; ++t) {
    const int ch = ((t & 1) ? 16 : 0) + l16;
    const float bias = (t < 2) ? lb[ch] : rb[ch];
    __bf16* dst = (t < 2) ? left_t : right_t;
#pragma unroll
    for (int v = 0; v < 8; ++v) {
      int row = r0 + 8 * hf + v;          // C/D layout: M = 8*(lane/16)+v
      float val = (acc[t][v] + bias) * mask[row];
      int b = row & (NRES - 1);
      int a = row >> 8;
      dst[(size_t)b * (CO * NSEQ) + ch * NSEQ + a] = (__bf16)val;
    }
  }
}

// ---------------------------------------------------------------------------
// Fused outer-product-mean + projection.
// One WG per 4x4 (b,d) tile: 128x128x512 bf16 GEMM (tmp kept in registers),
// right-tile K-chunks double-buffered in LDS via async global->LDS DMA,
// then reshape through LDS to [16 pairs x 1024] and x W2_t -> [16 x 128].
// Wave tiling: 2 M-subtiles x 4 N-subtiles per wave (halves LDS B reads).
// ---------------------------------------------------------------------------
#define RBUF_ELEMS (128 * 72)                 // one K-chunk buffer (bf16)
#define SMEM_BYTES (2 * RBUF_ELEMS * 2)       // 36864 B; s_ap (33024 B) overlays it

__global__ __launch_bounds__(256) void k_opm(
    const __bf16* __restrict__ left_t, const __bf16* __restrict__ right_t,
    const __bf16* __restrict__ w2t,    const float* __restrict__ nrm,
    const float* __restrict__ ob,      float* __restrict__ out) {
  __shared__ alignas(16) unsigned char smem[SMEM_BYTES];
  __bf16* s_right = (__bf16*)smem;            // 2 x [128 cols x 64 K], stride 72
  __bf16* s_ap    = (__bf16*)smem;            // [16 pairs x 1024], stride 1032 (after main loop)

  const int tid  = threadIdx.x;
  const int w    = tid >> 5;
  const int lane = tid & 31;
  const int hf   = lane >> 4;
  const int l16  = lane & 15;
  const int mw   = w >> 1;                    // M-group: subtiles 2*mw, 2*mw+1
  const int nw   = w & 1;                     // N-group: subtiles 4*nw .. 4*nw+3
  const int b0   = (blockIdx.x >> 6) * 4;
  const int d0   = (blockIdx.x & 63) * 4;

  v8f acc[2][4] = {};
  const __bf16* aRow[2];
#pragma unroll
  for (int mi = 0; mi < 2; ++mi) {
    int M = mw * 32 + mi * 16 + l16;          // bl = M>>5 == mw, c = M&31
    aRow[mi] = left_t + ((size_t)(b0 + mw) * CO + (M & 31)) * NSEQ;
  }

  // Async-issue one right-tile K-chunk (64 a-values, all 128 (d,e) cols) -> LDS buf bb.
  auto issue = [&](int kc, int bb) {
    int col = tid & 127, hh = tid >> 7;
    int dl = col >> 5, e = col & 31;
    const __bf16* src = right_t + ((size_t)(d0 + dl) * CO + e) * NSEQ + kc * 64 + hh * 32;
    unsigned dst = (unsigned)(size_t)(s_right + bb * RBUF_ELEMS + col * 72 + hh * 32);
#pragma unroll
    for (int i = 0; i < 4; ++i)
      asm volatile("global_load_async_to_lds_b128 %0, %1, off"
                   :: "v"(dst + i * 16), "v"(src + i * 8) : "memory");
  };

  issue(0, 0);
  for (int kc = 0; kc < 8; ++kc) {            // K chunks of 64 (K = a = 512)
    if (kc < 7) {
      issue(kc + 1, (kc + 1) & 1);            // buffer last read in the previous iteration
      __builtin_prefetch(aRow[0] + (kc + 1) * 64, 0, 1);
      __builtin_prefetch(aRow[1] + (kc + 1) * 64, 0, 1);
      asm volatile("s_wait_asynccnt 0x4" ::: "memory");   // chunk kc resident (in-order)
    } else {
      asm volatile("s_wait_asynccnt 0x0" ::: "memory");
    }
    __syncthreads();                          // all threads' chunk-kc writes visible
    const __bf16* sb = s_right + (kc & 1) * RBUF_ELEMS;
#pragma unroll
    for (int ks = 0; ks < 2; ++ks) {          // two k-steps of 32 per chunk
      const int k = kc * 64 + ks * 32;
      v16bf af[2];
#pragma unroll
      for (int mi = 0; mi < 2; ++mi) {
        const __bf16* ap = aRow[mi] + k + hf * 8;
        af[mi] = pack16(*(const v8bf*)ap, *(const v8bf*)(ap + 16));
      }
#pragma unroll
      for (int t = 0; t < 4; ++t) {
        const __bf16* bp = sb + ((nw * 4 + t) * 16 + l16) * 72 + ks * 32 + hf * 8;
        v16bf bfr = pack16(*(const v8bf*)bp, *(const v8bf*)(bp + 16));
#pragma unroll
        for (int mi = 0; mi < 2; ++mi)
          acc[mi][t] = __builtin_amdgcn_wmma_f32_16x16x32_bf16(
              false, af[mi], false, bfr, (short)0, acc[mi][t], false, false);
      }
    }
    __syncthreads();                          // reads done before buffer reuse/overlay
  }

  // Scatter tmp accumulators -> A' [pair p][ce] bf16 in LDS (overlays s_right).
#pragma unroll
  for (int t = 0; t < 4; ++t) {
    int N = nw * 64 + t * 16 + l16;
    int dl = N >> 5, e = N & 31;
#pragma unroll
    for (int mi = 0; mi < 2; ++mi)
#pragma unroll
      for (int v = 0; v < 8; ++v) {
        int cv = mi * 16 + 8 * hf + v;
        int p  = mw * 4 + dl;
        int ce = cv * 32 + e;
        s_ap[p * 1032 + ce] = (__bf16)acc[mi][t][v];
      }
  }
  __syncthreads();

  // Projection: [16 x 1024] @ [1024 x 128]; wave w owns f = 16w..16w+15.
  v8f po = {};
  const __bf16* wrow = w2t + (size_t)(w * 16 + l16) * 1024;
  for (int kk = 0; kk < 32; ++kk) {
    const int k = kk * 32;
    const __bf16* ap = s_ap + (size_t)l16 * 1032 + k + hf * 8;
    v16bf af = pack16(*(const v8bf*)ap, *(const v8bf*)(ap + 16));
    const __bf16* bp = wrow + k + hf * 8;
    v16bf bfr = pack16(*(const v8bf*)bp, *(const v8bf*)(bp + 16));
    po = __builtin_amdgcn_wmma_f32_16x16x32_bf16(
        false, af, false, bfr, (short)0, po, false, false);
  }

  // Epilogue: + output_b, / (eps + norm)
  const int f = w * 16 + l16;
  const float bias = ob[f];
#pragma unroll
  for (int v = 0; v < 8; ++v) {
    int p = 8 * hf + v;
    int b = b0 + (p >> 2);
    int d = d0 + (p & 3);
    float nm = nrm[b * NRES + d];
    out[((size_t)b * NRES + d) * CF + f] = (po[v] + bias) / (1e-3f + nm);
  }
}

// ---------------------------------------------------------------------------
extern "C" void kernel_launch(void* const* d_in, const int* in_sizes, int n_in,
                              void* d_out, int out_size, void* d_ws, size_t ws_size,
                              hipStream_t stream) {
  const float* act  = (const float*)d_in[0];
  const float* mask = (const float*)d_in[1];
  const float* ln_s = (const float*)d_in[2];
  const float* ln_o = (const float*)d_in[3];
  const float* lw   = (const float*)d_in[4];
  const float* lb   = (const float*)d_in[5];
  const float* rw   = (const float*)d_in[6];
  const float* rb   = (const float*)d_in[7];
  const float* ow   = (const float*)d_in[8];
  const float* ob   = (const float*)d_in[9];
  float* out = (float*)d_out;

  char* ws = (char*)d_ws;
  size_t off = 0;
  __bf16* left_t  = (__bf16*)(ws + off); off += (size_t)NRES * CO * NSEQ * 2;  // 8 MB
  __bf16* right_t = (__bf16*)(ws + off); off += (size_t)NRES * CO * NSEQ * 2;  // 8 MB
  __bf16* w2t     = (__bf16*)(ws + off); off += (size_t)CF * 1024 * 2;         // 256 KB
  float*  nrm     = (float*)(ws + off);  off += (size_t)NRES * NRES * 4;       // 256 KB

  k_w2t <<<512, 256, 0, stream>>>(ow, w2t);
  k_norm<<<NRES, 256, 0, stream>>>(mask, nrm);
  k_ln_lr<<<1024, 256, 0, stream>>>(act, mask, ln_s, ln_o, lw, lb, rw, rb, left_t, right_t);
  k_opm <<<64 * 64, 256, 0, stream>>>(left_t, right_t, w2t, nrm, ob, out);
}
